// LocalSelfAttention_5540507811946
// MI455X (gfx1250) — matrix operand order repaired and checked by reading, not compiled, not run on hardware
//
#include <hip/hip_runtime.h>
#include <hip/hip_bf16.h>
#include <stdint.h>

#define HH 56
#define WWID 56
#define HWP 3136      // 56*56
#define BATCH 4
#define NPIX 12544    // 4*3136
#define CIN 256
#define COUT 256
#define HDIM 128
#define PADR 2
#define EPSV 1e-5f
#define INV_SQRT_HD 0.08838834764831845f  // 1/sqrt(128)

typedef __attribute__((ext_vector_type(16))) _Float16 v16h;
typedef __attribute__((ext_vector_type(8)))  float    v8f;

union Frag16 { v16h v; float4 f[2]; };
union H8     { float4 f; _Float16 h[8]; };

// ---------------------------------------------------------------------------
// Convert x (B,C,H,W) f32 -> pixel-major f16  xt[p][c]   (p = b*HW + hw)
// ---------------------------------------------------------------------------
__global__ __launch_bounds__(256) void cvt_x_kernel(const float* __restrict__ x,
                                                    _Float16* __restrict__ xt) {
  int t = blockIdx.x * 256 + threadIdx.x;
  if (t >= BATCH * CIN * HWP) return;
  int hw = t % HWP;
  int c  = (t / HWP) % CIN;
  int b  = t / (HWP * CIN);
  xt[(size_t)(b * HWP + hw) * CIN + c] = (_Float16)x[t];
}

__global__ __launch_bounds__(256) void cvt_f16_kernel(const float* __restrict__ src,
                                                      _Float16* __restrict__ dst, int n) {
  int t = blockIdx.x * 256 + threadIdx.x;
  if (t < n) dst[t] = (_Float16)src[t];
}

// ---------------------------------------------------------------------------
// GEMM: Out16[p][m] = sum_c Wm[m][c] * X[p][c]   (M=256, K=256, N=NPIX)
// Block = 256 threads = 8 waves; block covers 32 pixels (2 N-tiles), each
// wave does 2 M-tiles x 2 N-tiles = 32 WMMAs. X tile staged in LDS via
// GLOBAL_LOAD_ASYNC_TO_LDS_B128 (ASYNCcnt) — gfx1250 async-copy path.
// ---------------------------------------------------------------------------
__global__ __launch_bounds__(256) void gemm256_kernel(
    const _Float16* __restrict__ Wm, const _Float16* __restrict__ X,
    _Float16* __restrict__ Out,
    const float* __restrict__ bn_g, const float* __restrict__ bn_b,
    const float* __restrict__ bn_m, const float* __restrict__ bn_v) {
  __shared__ __align__(32) _Float16 lx[32 * 272];   // 32 pixels x 256ch (+pad)
  const int t = threadIdx.x;
  const int pbase = blockIdx.x * 32;
  {
    // each thread async-copies 64B (4 x b128) of one pixel row into LDS
    int row = t >> 3;               // 0..31
    int col = (t & 7) * 32;         // half index
    uint64_t ga    = (uint64_t)(uintptr_t)(X + (size_t)(pbase + row) * CIN + col);
    uint32_t dsoff = (uint32_t)(uintptr_t)(lx + row * 272 + col);
#pragma unroll
    for (int i = 0; i < 4; ++i) {
      asm volatile("global_load_async_to_lds_b128 %0, %1, off"
                   :: "v"(dsoff + 16u * i), "v"(ga + 16u * i)
                   : "memory");
    }
  }
  asm volatile("s_wait_asynccnt 0x0" ::: "memory");
  __syncthreads();

  const int wave = t >> 5, lane = t & 31;
  const int n = lane & 15, g = lane >> 4;
  const int mt0 = wave * 2, mt1 = wave * 2 + 1;
  v8f acc00 = {}; v8f acc01 = {};   // (mt0, ntile0/1)
  v8f acc10 = {}; v8f acc11 = {};   // (mt1, ntile0/1)
  const _Float16* a0base = Wm + (size_t)(mt0 * 16 + n) * CIN + g * 8;
  const _Float16* a1base = Wm + (size_t)(mt1 * 16 + n) * CIN + g * 8;
  const _Float16* b0base = lx + n * 272 + g * 8;
  const _Float16* b1base = lx + (16 + n) * 272 + g * 8;
  __builtin_prefetch(a0base, 0, 0);
  __builtin_prefetch(a1base, 0, 0);
#pragma unroll
  for (int ks = 0; ks < 8; ++ks) {
    Frag16 b0, b1, a0, a1;
    b0.f[0] = *(const float4*)(b0base + ks * 32);
    b0.f[1] = *(const float4*)(b0base + ks * 32 + 16);
    b1.f[0] = *(const float4*)(b1base + ks * 32);
    b1.f[1] = *(const float4*)(b1base + ks * 32 + 16);
    a0.f[0] = *(const float4*)(a0base + ks * 32);
    a0.f[1] = *(const float4*)(a0base + ks * 32 + 16);
    a1.f[0] = *(const float4*)(a1base + ks * 32);
    a1.f[1] = *(const float4*)(a1base + ks * 32 + 16);
    acc00 = __builtin_amdgcn_wmma_f32_16x16x32_f16(false, a0.v, false, b0.v,
                                                   (short)0, acc00, false, false);
    acc01 = __builtin_amdgcn_wmma_f32_16x16x32_f16(false, a0.v, false, b1.v,
                                                   (short)0, acc01, false, false);
    acc10 = __builtin_amdgcn_wmma_f32_16x16x32_f16(false, a1.v, false, b0.v,
                                                   (short)0, acc10, false, false);
    acc11 = __builtin_amdgcn_wmma_f32_16x16x32_f16(false, a1.v, false, b1.v,
                                                   (short)0, acc11, false, false);
  }
#pragma unroll
  for (int sel = 0; sel < 4; ++sel) {
    v8f acc = (sel == 0) ? acc00 : (sel == 1) ? acc01 : (sel == 2) ? acc10 : acc11;
    int mt = (sel >> 1) ? mt1 : mt0;
    int pixel = pbase + ((sel & 1) ? 16 : 0) + n;
    int chbase = mt * 16 + g * 8;
    H8 o;
#pragma unroll
    for (int r = 0; r < 8; ++r) {
      float f = acc[r];
      if (bn_g) {
        int ch = chbase + r;
        float sc = bn_g[ch] * rsqrtf(bn_v[ch] + EPSV);
        f = f * sc + (bn_b[ch] - bn_m[ch] * sc);
      }
      o.h[r] = (_Float16)f;
    }
    *(float4*)(Out + (size_t)pixel * COUT + chbase) = o.f;
  }
}

// ---------------------------------------------------------------------------
// Local attention: one wave32 per pixel. Lane l owns channels [8l, 8l+8);
// lanes 0-15 = head 0, lanes 16-31 = head 1. Bias folded as q.rel dots.
// Epilogue: BN1 + ReLU, stored f16 pixel-major.
// ---------------------------------------------------------------------------
__global__ __launch_bounds__(256) void attn_kernel(
    const _Float16* __restrict__ Q, const _Float16* __restrict__ Kb,
    const _Float16* __restrict__ Vb,
    const float* __restrict__ rel_h, const float* __restrict__ rel_w,
    const float* __restrict__ g1, const float* __restrict__ b1,
    const float* __restrict__ m1, const float* __restrict__ v1,
    _Float16* __restrict__ AO) {
  const int wave = threadIdx.x >> 5, lane = threadIdx.x & 31;
  const int p = blockIdx.x * 8 + wave;
  const int b = p / HWP, hw = p % HWP;
  const int h = hw / WWID, w = hw % WWID;
  const int cb = lane * 8;
  const int head = lane >> 4;

  float qv[8];
  {
    H8 qh; qh.f = *(const float4*)(Q + (size_t)p * COUT + cb);
#pragma unroll
    for (int r = 0; r < 8; ++r) qv[r] = (float)qh.h[r];
  }

  // q . rel bias (5 taps per head); rel[c][t] at c*5+t
  const float* rel = head ? rel_w : rel_h;
  const int lc = (lane & 15) * 8;
  float rb[5];
#pragma unroll
  for (int i = 0; i < 5; ++i) {
    float s = 0.f;
#pragma unroll
    for (int r = 0; r < 8; ++r) s += qv[r] * rel[(lc + r) * 5 + i];
    for (int m = 1; m < 16; m <<= 1) s += __shfl_xor(s, m, 16);
    rb[i] = s;
  }

  float sc[25];
#pragma unroll
  for (int tap = 0; tap < 25; ++tap) {
    int di = tap / 5, dj = tap % 5;
    int hh = h + di - PADR, ww = w + dj - PADR;
    bool inb = ((unsigned)hh < HH) && ((unsigned)ww < WWID);
    float d = 0.f;
    if (inb) {
      int p2 = b * HWP + hh * WWID + ww;
      H8 kh; kh.f = *(const float4*)(Kb + (size_t)p2 * COUT + cb);
#pragma unroll
      for (int r = 0; r < 8; ++r) d += qv[r] * (float)kh.h[r];
    }
    for (int m = 1; m < 16; m <<= 1) d += __shfl_xor(d, m, 16);
    sc[tap] = (d + (head ? rb[dj] : rb[di])) * INV_SQRT_HD;
  }

  float mx = sc[0];
#pragma unroll
  for (int tap = 1; tap < 25; ++tap) mx = fmaxf(mx, sc[tap]);
  float sum = 0.f;
#pragma unroll
  for (int tap = 0; tap < 25; ++tap) { sc[tap] = __expf(sc[tap] - mx); sum += sc[tap]; }
  const float inv = 1.f / sum;

  float acc[8] = {0.f, 0.f, 0.f, 0.f, 0.f, 0.f, 0.f, 0.f};
#pragma unroll
  for (int tap = 0; tap < 25; ++tap) {
    int di = tap / 5, dj = tap % 5;
    int hh = h + di - PADR, ww = w + dj - PADR;
    if (((unsigned)hh < HH) && ((unsigned)ww < WWID)) {
      int p2 = b * HWP + hh * WWID + ww;
      H8 vh; vh.f = *(const float4*)(Vb + (size_t)p2 * COUT + cb);
      float a = sc[tap] * inv;
#pragma unroll
      for (int r = 0; r < 8; ++r) acc[r] += a * (float)vh.h[r];
    }
  }

  H8 o;
#pragma unroll
  for (int r = 0; r < 8; ++r) {
    int ch = cb + r;
    float s = g1[ch] * rsqrtf(v1[ch] + EPSV);
    float f = acc[r] * s + (b1[ch] - m1[ch] * s);
    o.h[r] = (_Float16)fmaxf(f, 0.f);
  }
  *(float4*)(AO + (size_t)p * COUT + cb) = o.f;
}

// ---------------------------------------------------------------------------
// SE projection: s[p][ch<16] = relu(BN(seWin @ y)), + per-(b,ch) sums for mean.
// One wave per 16-pixel tile, single 16x16 WMMA output tile (M=16).
// ---------------------------------------------------------------------------
__global__ __launch_bounds__(256) void se_kernel(
    const _Float16* __restrict__ Wse, const _Float16* __restrict__ Y,
    const float* __restrict__ g_in, const float* __restrict__ b_in,
    const float* __restrict__ m_in, const float* __restrict__ v_in,
    float* __restrict__ S, float* __restrict__ psum) {
  const int wave = threadIdx.x >> 5, lane = threadIdx.x & 31;
  const int ptile = blockIdx.x * 8 + wave;
  const int n = lane & 15, g = lane >> 4;
  const _Float16* abase = Wse + (size_t)n * CIN + g * 8;
  const _Float16* bbase = Y + (size_t)(ptile * 16 + n) * CIN + g * 8;
  v8f acc = {};
#pragma unroll
  for (int ks = 0; ks < 8; ++ks) {
    Frag16 a, bf;
    a.f[0]  = *(const float4*)(abase + ks * 32);
    a.f[1]  = *(const float4*)(abase + ks * 32 + 16);
    bf.f[0] = *(const float4*)(bbase + ks * 32);
    bf.f[1] = *(const float4*)(bbase + ks * 32 + 16);
    acc = __builtin_amdgcn_wmma_f32_16x16x32_f16(false, a.v, false, bf.v,
                                                 (short)0, acc, false, false);
  }
  const int pixel = ptile * 16 + n;
  const int bidx = (ptile * 16) / HWP;   // tiles never straddle batch (3136 % 16 == 0)
#pragma unroll
  for (int r = 0; r < 8; ++r) {
    int ch = r + g * 8;
    float s = g_in[ch] * rsqrtf(v_in[ch] + EPSV);
    float f = fmaxf(acc[r] * s + (b_in[ch] - m_in[ch] * s), 0.f);
    S[(size_t)pixel * 16 + ch] = f;
    float tt = f;
    for (int m = 1; m < 16; m <<= 1) tt += __shfl_xor(tt, m, 16);
    if (n == 0) atomicAdd(&psum[bidx * 16 + ch], tt);
  }
}

// g[b][ch] = sigmoid( relu(mean_p @ fc1^T) * fc2[ch] )
__global__ void gate_kernel(const float* __restrict__ psum,
                            const float* __restrict__ fc1,
                            const float* __restrict__ fc2,
                            float* __restrict__ G) {
  int t = threadIdx.x;
  if (t >= 64) return;
  int b = t >> 4, ch = t & 15;
  float acc = 0.f;
  for (int c = 0; c < 16; ++c) acc += (psum[b * 16 + c] * (1.0f / HWP)) * fc1[c];
  acc = fmaxf(acc, 0.f);
  G[t] = 1.f / (1.f + __expf(-acc * fc2[ch]));
}

// out[b][co][hw] = BN_out( sum_ch seWout[co][ch] * S[p][ch] * G[b][ch] )
__global__ __launch_bounds__(256) void final_kernel(
    const float* __restrict__ S, const float* __restrict__ G,
    const float* __restrict__ Wout,
    const float* __restrict__ g_o, const float* __restrict__ b_o,
    const float* __restrict__ m_o, const float* __restrict__ v_o,
    float* __restrict__ out) {
  int t = blockIdx.x * 256 + threadIdx.x;
  if (t >= BATCH * COUT * HWP) return;
  int hw = t % HWP;
  int co = (t / HWP) % COUT;
  int b  = t / (HWP * COUT);
  size_t p = (size_t)b * HWP + hw;
  float acc = 0.f;
#pragma unroll
  for (int ch = 0; ch < 16; ++ch)
    acc += Wout[co * 16 + ch] * S[p * 16 + ch] * G[b * 16 + ch];
  float s = g_o[co] * rsqrtf(v_o[co] + EPSV);
  out[t] = acc * s + (b_o[co] - m_o[co] * s);
}

// ---------------------------------------------------------------------------
extern "C" void kernel_launch(void* const* d_in, const int* in_sizes, int n_in,
                              void* d_out, int out_size, void* d_ws, size_t ws_size,
                              hipStream_t stream) {
  (void)in_sizes; (void)n_in; (void)out_size; (void)ws_size;
  const float* x     = (const float*)d_in[0];
  const float* Wq    = (const float*)d_in[1];
  const float* Wk    = (const float*)d_in[2];
  const float* Wv    = (const float*)d_in[3];
  const float* rel_h = (const float*)d_in[4];
  const float* rel_w = (const float*)d_in[5];
  const float* g1    = (const float*)d_in[6];
  const float* b1    = (const float*)d_in[7];
  const float* m1    = (const float*)d_in[8];
  const float* v1    = (const float*)d_in[9];
  const float* aggW  = (const float*)d_in[10];
  const float* g2    = (const float*)d_in[11];
  const float* b2    = (const float*)d_in[12];
  const float* m2    = (const float*)d_in[13];
  const float* v2    = (const float*)d_in[14];
  const float* seWin = (const float*)d_in[15];
  const float* g_in  = (const float*)d_in[16];
  const float* b_in  = (const float*)d_in[17];
  const float* m_in  = (const float*)d_in[18];
  const float* v_in  = (const float*)d_in[19];
  const float* fc1   = (const float*)d_in[20];
  const float* fc2   = (const float*)d_in[21];
  const float* seWout= (const float*)d_in[22];
  const float* g_o   = (const float*)d_in[23];
  const float* b_o   = (const float*)d_in[24];
  const float* m_o   = (const float*)d_in[25];
  const float* v_o   = (const float*)d_in[26];

  uint8_t* ws = (uint8_t*)d_ws;
  size_t off = 0;
  auto alloc = [&](size_t bytes) -> void* {
    void* p = ws + off;
    off = (off + bytes + 255) & ~(size_t)255;
    return p;
  };

  const size_t big = (size_t)NPIX * CIN * sizeof(_Float16);   // 6.4 MB
  _Float16* XT16   = (_Float16*)alloc(big);   // x^T f16; later reused as attn-out (AO)
  _Float16* Q16    = (_Float16*)alloc(big);   // q f16;  later reused as y (agg output)
  _Float16* K16    = (_Float16*)alloc(big);
  _Float16* V16    = (_Float16*)alloc(big);
  _Float16* Wq16   = (_Float16*)alloc((size_t)COUT * CIN * 2);
  _Float16* Wk16   = (_Float16*)alloc((size_t)COUT * CIN * 2);
  _Float16* Wv16   = (_Float16*)alloc((size_t)COUT * CIN * 2);
  _Float16* aggW16 = (_Float16*)alloc((size_t)COUT * CIN * 2);
  _Float16* seWin16= (_Float16*)alloc((size_t)16 * CIN * 2);
  float* S  = (float*)alloc((size_t)NPIX * 16 * sizeof(float));
  float* PS = (float*)alloc(64 * sizeof(float));
  float* G  = (float*)alloc(64 * sizeof(float));

  // 1) precision conversions / layout
  cvt_x_kernel<<<(BATCH * CIN * HWP) / 256, 256, 0, stream>>>(x, XT16);
  cvt_f16_kernel<<<256, 256, 0, stream>>>(Wq,   Wq16,   COUT * CIN);
  cvt_f16_kernel<<<256, 256, 0, stream>>>(Wk,   Wk16,   COUT * CIN);
  cvt_f16_kernel<<<256, 256, 0, stream>>>(Wv,   Wv16,   COUT * CIN);
  cvt_f16_kernel<<<256, 256, 0, stream>>>(aggW, aggW16, COUT * CIN);
  cvt_f16_kernel<<<16,  256, 0, stream>>>(seWin, seWin16, 16 * CIN);

  // 2) q/k/v projections (WMMA GEMMs, 32 pixels per block)
  gemm256_kernel<<<NPIX / 32, 256, 0, stream>>>(Wq16, XT16, Q16,
                                                nullptr, nullptr, nullptr, nullptr);
  gemm256_kernel<<<NPIX / 32, 256, 0, stream>>>(Wk16, XT16, K16,
                                                nullptr, nullptr, nullptr, nullptr);
  gemm256_kernel<<<NPIX / 32, 256, 0, stream>>>(Wv16, XT16, V16,
                                                nullptr, nullptr, nullptr, nullptr);

  // 3) local windowed attention + BN1 + ReLU  (writes AO into XT16 buffer)
  attn_kernel<<<NPIX / 8, 256, 0, stream>>>(Q16, K16, V16, rel_h, rel_w,
                                            g1, b1, m1, v1, XT16);

  // 4) agg 1x1 conv + BN2  (WMMA GEMM; y written into Q16 buffer)
  gemm256_kernel<<<NPIX / 32, 256, 0, stream>>>(aggW16, XT16, Q16, g2, b2, m2, v2);

  // 5) SE: projection + spatial mean, gate, final conv + BN_out
  hipMemsetAsync(PS, 0, 64 * sizeof(float), stream);
  se_kernel<<<NPIX / 16 / 8, 256, 0, stream>>>(seWin16, Q16, g_in, b_in, m_in, v_in,
                                               S, PS);
  gate_kernel<<<1, 64, 0, stream>>>(PS, fc1, fc2, G);
  final_kernel<<<(BATCH * COUT * HWP) / 256, 256, 0, stream>>>(
      S, G, seWout, g_o, b_o, m_o, v_o, (float*)d_out);
}
